// BiDiTreeLSTM_19172734010036
// MI455X (gfx1250) — matrix operand order repaired and checked by reading, not compiled, not run on hardware
//
#include <hip/hip_runtime.h>
#include <hip/hip_bf16.h>
#include <math.h>

// ---------------------------------------------------------------------------
// Types for CDNA5 WMMA (wave32): v_wmma_f32_16x16x32_bf16
//   A: 16x32 bf16 (16 elems/lane), B: 32x16 bf16 (16 elems/lane), C/D: v8f
// ---------------------------------------------------------------------------
typedef __attribute__((ext_vector_type(16))) __bf16 v16bf;
typedef __attribute__((ext_vector_type(8)))  __bf16 v8bf;
typedef __attribute__((ext_vector_type(8)))  float  v8f;

union ABFrag { v16bf v; v8bf h[2]; };

// Force global (addrspace(1)) memory ops so the backend emits
// global_load_b128 / global_store_b32 with LOADcnt/STOREcnt-only waits,
// instead of flat_* which also charges DScnt and forces full drains.
typedef const __attribute__((address_space(1))) v8bf*  gv8bf_cp;
typedef __attribute__((address_space(1))) float*       gf32_p;

__device__ __forceinline__ v8bf gload8(const __bf16* p) {
  return *(gv8bf_cp)(unsigned long long)p;
}
__device__ __forceinline__ void gstoref(float* p, float v) {
  *(gf32_p)(unsigned long long)p = v;
}

// A GEMM "segment": C += A_seg @ B_seg^T over K columns.
// A rows are tree-node rows: row r -> (b, j) -> node = nodeBase + ((j*mul)>>shr)
// A row pointer = A + (b*Ntot + node)*rowElems. B is [out][K] row-major bf16.
struct Seg {
  const __bf16* A;
  const __bf16* Bw;
  int K;        // multiple of 32
  int rowElems; // A row stride in elements
  int nodeBase;
  int nodeMul;  // 1 or 2
  int nodeShr;  // 0 or 1
};

// ---------------------------------------------------------------------------
// Fused multi-segment WMMA GEMM.
// Block = 128 threads = 4 waves. Block tile: 32 rows x 256 cols.
// Each wave owns 32x64 (2 row frags x 4 col tiles = 8 WMMA accumulators),
// accumulating all segments in registers. Per k-step: 4 A-b128 + 8 B-b128
// loads feed 8 WMMAs (1.5 loads/WMMA). grid = (M/32, Ncols/256).
// ---------------------------------------------------------------------------
template <int NS>
__global__ __launch_bounds__(128) void gemm_wmma_bf16(
    float* __restrict__ C, int ldc, int nlBits, int Ntot,
    Seg s0, Seg s1, Seg s2)
{
  Seg segs[3] = {s0, s1, s2};
  const int lane    = threadIdx.x & 31;
  const int wave    = threadIdx.x >> 5;
  const int colBase = blockIdx.y * 256 + wave * 64;
  const int rowTop  = blockIdx.x * 32;
  const int m0      = rowTop + (lane & 15);   // rows for A frag 0
  const int m1      = m0 + 16;                // rows for A frag 1
  const int half    = lane >> 4;              // lane-half select (ISA layout)
  const int mask    = (1 << nlBits) - 1;

  v8f acc[8];
  const v8f vz = {0.f, 0.f, 0.f, 0.f, 0.f, 0.f, 0.f, 0.f};
#pragma unroll
  for (int t = 0; t < 8; ++t) acc[t] = vz;

#pragma unroll
  for (int s = 0; s < NS; ++s) {
    const Seg sg = segs[s];
    const int b0 = m0 >> nlBits, j0 = m0 & mask;
    const int b1 = m1 >> nlBits, j1 = m1 & mask;
    const int node0 = sg.nodeBase + ((j0 * sg.nodeMul) >> sg.nodeShr);
    const int node1 = sg.nodeBase + ((j1 * sg.nodeMul) >> sg.nodeShr);
    // A fragment: lanes 0-15 hold K=[0..7 | 16..23], lanes 16-31 K=[8..15 | 24..31]
    const __bf16* Arow0 = sg.A + (size_t)(b0 * Ntot + node0) * sg.rowElems + half * 8;
    const __bf16* Arow1 = sg.A + (size_t)(b1 * Ntot + node1) * sg.rowElems + half * 8;

    for (int k = 0; k < sg.K; k += 32) {
      ABFrag a0, a1;
      a0.h[0] = gload8(Arow0 + k);
      a0.h[1] = gload8(Arow0 + k + 16);
      a1.h[0] = gload8(Arow1 + k);
      a1.h[1] = gload8(Arow1 + k + 16);
      ABFrag bf[4];
#pragma unroll
      for (int ct = 0; ct < 4; ++ct) {
        // B fragment: lane = column N; lanes 0-15 K=[0..7 | 16..23], lanes 16-31 K=[8..15 | 24..31]
        const int brow = colBase + ct * 16 + (lane & 15);
        const __bf16* Bp = sg.Bw + (size_t)brow * sg.K + k + half * 8;
        bf[ct].h[0] = gload8(Bp);
        bf[ct].h[1] = gload8(Bp + 16);
      }
#pragma unroll
      for (int ct = 0; ct < 4; ++ct) {
        acc[ct] = __builtin_amdgcn_wmma_f32_16x16x32_bf16(
            false, a0.v, false, bf[ct].v, (short)0, acc[ct], false, false);
        acc[4 + ct] = __builtin_amdgcn_wmma_f32_16x16x32_bf16(
            false, a1.v, false, bf[ct].v, (short)0, acc[4 + ct], false, false);
      }
    }
  }

  // C/D layout: VGPR r -> row (r + half*8), lane&15 -> column
#pragma unroll
  for (int g = 0; g < 2; ++g) {
#pragma unroll
    for (int ct = 0; ct < 4; ++ct) {
#pragma unroll
      for (int r = 0; r < 8; ++r) {
        size_t off = (size_t)(rowTop + g * 16 + r + half * 8) * ldc
                   + colBase + ct * 16 + (lane & 15);
        gstoref(C + off, acc[g * 4 + ct][r]);
      }
    }
  }
}

// ---------------------------------------------------------------------------
// Elementwise helpers / gate kernels
// ---------------------------------------------------------------------------
__device__ __forceinline__ float sigmf(float x) { return 1.f / (1.f + __expf(-x)); }

__global__ __launch_bounds__(256) void f32_to_bf16_k(
    const float* __restrict__ src, __bf16* __restrict__ dst, size_t n)
{
  size_t i = (size_t)blockIdx.x * blockDim.x + threadIdx.x;
  if (i < n) dst[i] = (__bf16)src[i];
}

// Bottom-up gates: one block per node-row r, thread = h index t.
__global__ __launch_bounds__(256) void gates_bu_k(
    const float* __restrict__ iou, const float* __restrict__ fpre,
    const float* __restrict__ c0,
    const float* __restrict__ b_iou, const float* __restrict__ b_f,
    float* __restrict__ h, float* __restrict__ c, __bf16* __restrict__ hbf,
    int levBits, int lo, int clo, int isLeaf)
{
  const int r = blockIdx.x, t = threadIdx.x;
  const int b = r >> levBits;
  const int j = r & ((1 << levBits) - 1);
  const int node = lo + j;
  const size_t gi = ((size_t)(b * 1023 + node)) * 256 + t;
  const size_t ib = (size_t)r * 768 + t;
  const float i_ = iou[ib]       + b_iou[t];
  const float o_ = iou[ib + 256] + b_iou[256 + t];
  const float u_ = iou[ib + 512] + b_iou[512 + t];
  float cred;
  if (isLeaf) {
    cred = c0[gi];
  } else {
    const int rr = (b << (levBits + 1)) + 2 * j;
    const float f0 = sigmf(fpre[(size_t)rr * 256 + t] + b_f[t]);
    const float f1 = sigmf(fpre[(size_t)(rr + 1) * 256 + t] + b_f[t]);
    const size_t ci = ((size_t)(b * 1023 + clo + 2 * j)) * 256 + t;
    cred = f0 * c[ci] + f1 * c[ci + 256];
  }
  const float cc = sigmf(i_) * tanhf(u_) + cred;
  const float hh = sigmf(o_) * tanhf(cc);
  c[gi] = cc; h[gi] = hh; hbf[gi] = (__bf16)hh;
}

// Top-down gates.
__global__ __launch_bounds__(256) void gates_td_k(
    const float* __restrict__ iou, const float* __restrict__ fpre,
    const float* __restrict__ c0,
    const float* __restrict__ b_iou, const float* __restrict__ b_f,
    float* __restrict__ h, float* __restrict__ c, __bf16* __restrict__ hbf,
    int levBits, int lo, int plo, int isRoot)
{
  const int r = blockIdx.x, t = threadIdx.x;
  const int b = r >> levBits;
  const int j = r & ((1 << levBits) - 1);
  const int node = lo + j;
  const size_t gi = ((size_t)(b * 1023 + node)) * 256 + t;
  const size_t ib = (size_t)r * 768 + t;
  const float i_ = iou[ib]       + b_iou[t];
  const float o_ = iou[ib + 256] + b_iou[256 + t];
  const float u_ = iou[ib + 512] + b_iou[512 + t];
  float cred;
  if (isRoot) {
    cred = c0[gi];
  } else {
    const float f = sigmf(fpre[(size_t)r * 256 + t] + b_f[t]);
    const int parent = plo + (j >> 1);
    cred = f * c[((size_t)(b * 1023 + parent)) * 256 + t];
  }
  const float cc = sigmf(i_) * tanhf(u_) + cred;
  const float hh = sigmf(o_) * tanhf(cc);
  c[gi] = cc; h[gi] = hh; hbf[gi] = (__bf16)hh;
}

// X_td = concat([X, h_bu], -1) in bf16. n = 64*1023*512.
__global__ __launch_bounds__(256) void concat_xtd_k(
    const float* __restrict__ X, const float* __restrict__ h,
    __bf16* __restrict__ xtd, size_t n)
{
  size_t i = (size_t)blockIdx.x * blockDim.x + threadIdx.x;
  if (i >= n) return;
  size_t bn = i >> 9;
  int t = (int)(i & 511);
  float v = (t < 256) ? X[bn * 256 + t] : h[bn * 256 + (t - 256)];
  xtd[i] = (__bf16)v;
}

__global__ __launch_bounds__(256) void copy_root_k(
    const float* __restrict__ h, float* __restrict__ out)
{
  const int b = blockIdx.x, t = threadIdx.x;
  out[(size_t)b * 512 + t] = h[((size_t)b * 1023) * 256 + t];
}

__global__ __launch_bounds__(256) void leaf_mean_k(
    const float* __restrict__ h, float* __restrict__ out)
{
  const int b = blockIdx.x, t = threadIdx.x;
  float s = 0.f;
  for (int j = 0; j < 512; ++j)
    s += h[((size_t)(b * 1023 + 511 + j)) * 256 + t];
  out[(size_t)b * 512 + 256 + t] = s * (1.0f / 512.0f);
}

// ---------------------------------------------------------------------------
// Host-side orchestration
// ---------------------------------------------------------------------------
static inline void* wsAlloc(uint8_t*& p, size_t bytes) {
  void* r = (void*)p;
  p += (bytes + 255) & ~(size_t)255;
  return r;
}

extern "C" void kernel_launch(void* const* d_in, const int* in_sizes, int n_in,
                              void* d_out, int out_size, void* d_ws, size_t ws_size,
                              hipStream_t stream) {
  (void)in_sizes; (void)n_in; (void)out_size; (void)ws_size;
  const float* X      = (const float*)d_in[0];
  const float* c0     = (const float*)d_in[2];
  const float* Wbu    = (const float*)d_in[3];
  const float* Ubu    = (const float*)d_in[4];
  const float* b_bu   = (const float*)d_in[5];
  const float* Ufbu   = (const float*)d_in[6];
  const float* bf_bu  = (const float*)d_in[7];
  const float* Wtd    = (const float*)d_in[8];
  const float* Utd    = (const float*)d_in[9];
  const float* b_td   = (const float*)d_in[10];
  const float* Uftd   = (const float*)d_in[11];
  const float* bf_td  = (const float*)d_in[12];
  float* out = (float*)d_out;

  const int Bsz = 64, Ntot = 1023, depth = 9;
  const size_t nXH = (size_t)Bsz * Ntot * 256;

  uint8_t* wp = (uint8_t*)d_ws;
  __bf16* Xbf   = (__bf16*)wsAlloc(wp, nXH * 2);
  __bf16* XTDbf = (__bf16*)wsAlloc(wp, (size_t)Bsz * Ntot * 512 * 2);
  float*  h     = (float*) wsAlloc(wp, nXH * 4);
  float*  c     = (float*) wsAlloc(wp, nXH * 4);
  __bf16* hbf   = (__bf16*)wsAlloc(wp, nXH * 2);
  float*  iou   = (float*) wsAlloc(wp, (size_t)32768 * 768 * 4);
  float*  fpre  = (float*) wsAlloc(wp, (size_t)32768 * 256 * 4);
  __bf16* Wbu_b = (__bf16*)wsAlloc(wp, (size_t)768 * 256 * 2);
  __bf16* Ubu_b = (__bf16*)wsAlloc(wp, (size_t)768 * 256 * 2);
  __bf16* Ufbu_b= (__bf16*)wsAlloc(wp, (size_t)256 * 256 * 2);
  __bf16* Wtd_b = (__bf16*)wsAlloc(wp, (size_t)768 * 512 * 2);
  __bf16* Utd_b = (__bf16*)wsAlloc(wp, (size_t)768 * 256 * 2);
  __bf16* Uftd_b= (__bf16*)wsAlloc(wp, (size_t)256 * 256 * 2);

  auto cvt = [&](const float* s, __bf16* d, size_t n) {
    f32_to_bf16_k<<<(unsigned)((n + 255) / 256), 256, 0, stream>>>(s, d, n);
  };
  cvt(X, Xbf, nXH);
  cvt(Wbu,  Wbu_b,  (size_t)768 * 256);
  cvt(Ubu,  Ubu_b,  (size_t)768 * 256);
  cvt(Ufbu, Ufbu_b, (size_t)256 * 256);
  cvt(Wtd,  Wtd_b,  (size_t)768 * 512);
  cvt(Utd,  Utd_b,  (size_t)768 * 256);
  cvt(Uftd, Uftd_b, (size_t)256 * 256);

  auto gemm = [&](float* Cp, int ldc, int nlBits, dim3 grid,
                  int ns, Seg a, Seg b, Seg cc) {
    switch (ns) {
      case 1: gemm_wmma_bf16<1><<<grid, 128, 0, stream>>>(Cp, ldc, nlBits, Ntot, a, b, cc); break;
      case 2: gemm_wmma_bf16<2><<<grid, 128, 0, stream>>>(Cp, ldc, nlBits, Ntot, a, b, cc); break;
      default: gemm_wmma_bf16<3><<<grid, 128, 0, stream>>>(Cp, ldc, nlBits, Ntot, a, b, cc); break;
    }
  };

  // ---------------- Bottom-up ----------------
  for (int lev = depth; lev >= 0; --lev) {
    const int nl = 1 << lev, lo = nl - 1, clo = 2 * lo + 1;
    const int M = Bsz * nl;
    Seg sX = {Xbf, Wbu_b, 256, 256, lo, 1, 0};
    if (lev == depth) {
      gemm(iou, 768, lev, dim3(M / 32, 3), 1, sX, sX, sX);
    } else {
      Seg sE = {hbf, Ubu_b, 256, 256, clo,     2, 0};
      Seg sO = {hbf, Ubu_b, 256, 256, clo + 1, 2, 0};
      gemm(iou, 768, lev, dim3(M / 32, 3), 3, sX, sE, sO);
      Seg sF = {hbf, Ufbu_b, 256, 256, clo, 1, 0};
      gemm(fpre, 256, lev + 1, dim3((2 * M) / 32, 1), 1, sF, sF, sF);
    }
    gates_bu_k<<<M, 256, 0, stream>>>(iou, fpre, c0, b_bu, bf_bu,
                                      h, c, hbf, lev, lo, clo, lev == depth);
  }

  copy_root_k<<<Bsz, 256, 0, stream>>>(h, out);
  {
    size_t n = (size_t)Bsz * Ntot * 512;
    concat_xtd_k<<<(unsigned)((n + 255) / 256), 256, 0, stream>>>(X, h, XTDbf, n);
  }

  // ---------------- Top-down ----------------
  for (int lev = 0; lev <= depth; ++lev) {
    const int nl = 1 << lev, lo = nl - 1;
    const int plo = (lev > 0) ? ((1 << (lev - 1)) - 1) : 0;
    const int M = Bsz * nl;
    Seg sX = {XTDbf, Wtd_b, 512, 512, lo, 1, 0};
    if (lev == 0) {
      gemm(iou, 768, lev, dim3(M / 32, 3), 1, sX, sX, sX);
    } else {
      Seg sP = {hbf, Utd_b, 256, 256, plo, 1, 1};
      gemm(iou, 768, lev, dim3(M / 32, 3), 2, sX, sP, sP);
      Seg sPF = {hbf, Uftd_b, 256, 256, plo, 1, 1};
      gemm(fpre, 256, lev, dim3(M / 32, 1), 1, sPF, sPF, sPF);
    }
    gates_td_k<<<M, 256, 0, stream>>>(iou, fpre, c0, b_td, bf_td,
                                      h, c, hbf, lev, lo, plo, lev == 0);
  }

  leaf_mean_k<<<Bsz, 256, 0, stream>>>(h, out);
}
